// LocationEncoder_76656576299537
// MI455X (gfx1250) — compile-verified, matrix operand order checked by reading, not run on hardware
//
#include <hip/hip_runtime.h>

// ---------------------------------------------------------------------------
// MI455X (gfx1250) fused retrieval pipeline:
//   k0a: queries f32 -> bf16 (ws)
//   k0b: DB f32 -> bf16 (ws, 102.4MB) -- paid once, enables TDM tile loads
//   k1 : TDM (tensor_load_to_lds) double-buffered DB tiles (2 x 132KB LDS,
//        hardware row padding -> bank-conflict-free), v_wmma_f32_16x16x32_bf16
//        simT = DB*Q^T with explicit dual A-fragment pipelining,
//        per-lane u64-key top-16/query in VGPRs (v_min/max_u64 inserts),
//        ds_swizzle lane-pair merge, per-(query,chunk) key lists -> ws
//   k2 : reduce 3136 keys/query -> final top-16 indices
//   k3 : gather high-res rows + mean-pool -> out
// ---------------------------------------------------------------------------

#define NB    1024        // queries
#define DSAT  256         // satclip dim (GEMM K)
#define NDB   200000      // database rows
#define DHI   1024        // high-res dim
#define KSEL  16          // top-k
#define SUB   256         // DB rows per TDM tile
#define NSUBS 4           // tiles per workgroup chunk (1024 rows)
#define NCHNK 196         // chunks: 196*1024 = 200704 >= NDB
#define RSTRIDE 264       // padded LDS row stride in bf16 elems (528B: 132%64=4 banks)

typedef __attribute__((ext_vector_type(16))) __bf16        v16bf;
typedef __attribute__((ext_vector_type(8)))  float         v8f;
typedef __attribute__((ext_vector_type(4)))  unsigned int  v4u;
typedef __attribute__((ext_vector_type(8)))  int           v8i;
typedef __attribute__((ext_vector_type(4)))  int           v4i;

union FragBF { v16bf v; unsigned int u[8]; uint4 q[2]; };
union AccF   { v8f   v; float f[8]; };

static __device__ __forceinline__ unsigned int f32_to_bf16(float f) {
  unsigned int u = __float_as_uint(f);
  return (u + 0x7FFFu + ((u >> 16) & 1u)) >> 16;   // round-to-nearest-even
}

// Monotonic float->u32 map: order of ku matches order of f.
static __device__ __forceinline__ unsigned int f32_key(float f) {
  unsigned int u = __float_as_uint(f);
  return u ^ ((unsigned int)((int)u >> 31) | 0x80000000u);
}

// Sorted-descending u64-key top-k; position-independent per-slot selects lower
// to v_min_u64/v_max_u64 chains behind a v_cmpx + execz skip.
static __device__ __forceinline__ void topk_insert_key(unsigned long long (&tk)[KSEL],
                                                       unsigned long long key) {
  if (key > tk[KSEL - 1]) {
    unsigned long long nt[KSEL];
    nt[0] = tk[0] > key ? tk[0] : key;
#pragma unroll
    for (int j = 1; j < KSEL; ++j)
      nt[j] = tk[j] > key ? tk[j] : (tk[j - 1] > key ? key : tk[j - 1]);
#pragma unroll
    for (int j = 0; j < KSEL; ++j) tk[j] = nt[j];
  }
}

// ---------------- k0a: queries f32 -> bf16 ----------------
__global__ __launch_bounds__(256) void convert_queries_kernel(
    const float* __restrict__ A, unsigned int* __restrict__ qbf) {
  const int i = blockIdx.x * 256 + threadIdx.x;           // NB*DSAT/2 pairs
  const float2 f = ((const float2*)A)[i];
  qbf[i] = (f32_to_bf16(f.y) << 16) | f32_to_bf16(f.x);
}

// ---------------- k0b: DB f32 -> bf16 ----------------
__global__ __launch_bounds__(256) void convert_db_kernel(
    const float* __restrict__ db, unsigned int* __restrict__ dbbf) {
  const int i = blockIdx.x * 256 + threadIdx.x;           // NDB*DSAT/4 quads
  const float4 f = ((const float4*)db)[i];
  const unsigned int p0 = (f32_to_bf16(f.y) << 16) | f32_to_bf16(f.x);
  const unsigned int p1 = (f32_to_bf16(f.w) << 16) | f32_to_bf16(f.z);
  ((uint2*)dbbf)[i] = make_uint2(p0, p1);
}

// ---------------- k1: TDM-fed WMMA GEMM (simT = DB * Q^T) + fused top-16 ----------------
// grid = (2, NCHNK): x = query half (512 q), y = DB chunk (1024 rows).
__global__ __launch_bounds__(256) void gemm_topk_kernel(
    const unsigned int* __restrict__ dbbf, const unsigned int* __restrict__ qbf,
    unsigned long long* __restrict__ candK) {
  __shared__ __align__(16) unsigned int sdb_u[2][SUB * RSTRIDE / 2];  // 2 x 132KB

  const int g    = blockIdx.y;
  const int yq   = blockIdx.x;
  const int tid  = threadIdx.x;
  const int w    = tid >> 5;
  const int lane = tid & 31;
  const int half = lane >> 4;
  const uint4* qb4 = (const uint4*)qbf;
  const unsigned long long dbbase = (unsigned long long)(uintptr_t)dbbf;

  // Issue one Tensor-Data-Mover DMA: 256x256 bf16 tile, hardware-padded rows
  // (pad_interval=6 -> every 512B, pad_amount=3 -> +16B) => 528B LDS row stride.
  auto issue_tdm = [&](int buf, int s) {
    const int row0 = g * (NSUBS * SUB) + s * SUB;
    const long long rem = (long long)NDB - row0;
    const unsigned int tdim1 = rem > 0 ? (unsigned int)rem : 0u;  // OOB rows read 0
    const unsigned long long ga = dbbase + (unsigned long long)row0 * (DSAT * 2);
    const unsigned int lds = (unsigned int)(uintptr_t)(&sdb_u[buf][0]);

    v4u g0;
    g0[0] = 1u;                                            // count=1, user desc
    g0[1] = lds;                                           // lds_addr (bytes)
    g0[2] = (unsigned int)(ga & 0xFFFFFFFFull);            // global_addr[31:0]
    g0[3] = ((unsigned int)(ga >> 32) & 0x01FFFFFFu) | (2u << 30);  // addr[56:32]|type=2

    v8i g1;
    g1[0] = (int)((1u << 16) | (1u << 20) | (6u << 22) | (3u << 25)); // dsz=2B,pad
    g1[1] = (int)(256u << 16);                             // tensor_dim0=256 (lo16)
    g1[2] = (int)((tdim1 & 0xFFFFu) << 16);                // dim0 hi=0, dim1 lo16
    g1[3] = (int)(((tdim1 >> 16) & 0xFFFFu) | (256u << 16)); // dim1 hi16, tile_dim0=256
    g1[4] = (int)256u;                                     // tile_dim1=256, tile_dim2=0
    g1[5] = (int)256u;                                     // tensor_dim0_stride=256
    g1[6] = 0;                                             // stride hi, dim1_stride lo
    g1[7] = 0;

    const v4i gz4 = {0, 0, 0, 0};                          // groups 2/3 unused (2D)
    const v8i gz8 = {0, 0, 0, 0, 0, 0, 0, 0};
    __builtin_amdgcn_tensor_load_to_lds(g0, g1, gz4, gz4, gz8, 0);
  };

  if (w == 0) {
    issue_tdm(0, 0);
    __builtin_amdgcn_s_wait_tensorcnt(0);
  }
  __syncthreads();

  for (int s = 0; s < NSUBS; ++s) {
    if (w == 0 && s + 1 < NSUBS) issue_tdm((s + 1) & 1, s + 1);  // overlap DMA
    const unsigned int* sdb = sdb_u[s & 1];
    const int row0 = g * (NSUBS * SUB) + s * SUB;

    // Each wave owns a fixed 16-query group per pass (4 x 8 waves x 16 = 512 q).
    for (int pass = 0; pass < 4; ++pass) {
      const int q = yq * 512 + pass * 128 + w * 16 + (lane & 15);

      // B fragments (queries) for K=256: lane<16 holds K{0..7,16..23}, else {8..15,24..31}.
      FragBF bfr[8];
#pragma unroll
      for (int kf = 0; kf < 8; ++kf) {
        const int e0 = q * DSAT + kf * 32 + (half << 3);
        bfr[kf].q[0] = qb4[e0 >> 3];
        bfr[kf].q[1] = qb4[(e0 >> 3) + 2];
      }

      unsigned long long tk[KSEL];
#pragma unroll
      for (int j = 0; j < KSEL; ++j) tk[j] = 0ull;

      for (int mt = 0; mt < SUB / 16; ++mt) {
        const int abase = (mt * 16 + (lane & 15)) * (RSTRIDE / 2) + (half << 2);

        AccF acc;
#pragma unroll
        for (int j = 0; j < 8; ++j) acc.f[j] = 0.f;

        // Explicit dual-buffer A-fragment pipeline (no inter-buffer copies, so
        // the allocator cannot coalesce them): while WMMA consumes one buffer,
        // the ds_load pair for the next kf is already in flight.
        FragBF a0, a1;
        a0.q[0] = *(const uint4*)&sdb[abase];
        a0.q[1] = *(const uint4*)&sdb[abase + 8];
#pragma unroll
        for (int kp = 0; kp < 4; ++kp) {
          {
            const int e1 = abase + (2 * kp + 1) * 16;
            a1.q[0] = *(const uint4*)&sdb[e1];
            a1.q[1] = *(const uint4*)&sdb[e1 + 8];
          }
          acc.v = __builtin_amdgcn_wmma_f32_16x16x32_bf16(
              false, a0.v, false, bfr[2 * kp].v, (short)0, acc.v, false, false);
          if (kp < 3) {
            const int e2 = abase + (2 * kp + 2) * 16;
            a0.q[0] = *(const uint4*)&sdb[e2];
            a0.q[1] = *(const uint4*)&sdb[e2 + 8];
          }
          acc.v = __builtin_amdgcn_wmma_f32_16x16x32_bf16(
              false, a1.v, false, bfr[2 * kp + 1].v, (short)0, acc.v, false, false);
        }

        // D layout: lane owns query col N=lane%16; VGPR j holds DB row j+8*half.
        const int rbase = row0 + mt * 16 + (half << 3);
#pragma unroll
        for (int j = 0; j < 8; ++j) {
          const int idx = rbase + j;
          unsigned int ku = f32_key(acc.f[j]);
          ku = (idx < NDB) ? ku : 0u;                       // mask tail rows
          const unsigned long long key =
              ((unsigned long long)ku << 32) | (unsigned int)idx;
          topk_insert_key(tk, key);
        }
      }

      // Merge partner lane (l and l^16 serve the same query) via SWAPX16 swizzle.
#pragma unroll
      for (int j = 0; j < KSEL; ++j) {
        const int plo = __builtin_amdgcn_ds_swizzle((int)(tk[j] & 0xFFFFFFFFull), 0x401F);
        const int phi = __builtin_amdgcn_ds_swizzle((int)(tk[j] >> 32), 0x401F);
        const unsigned long long pk =
            ((unsigned long long)(unsigned int)phi << 32) | (unsigned int)plo;
        if (lane < 16) topk_insert_key(tk, pk);
      }

      // Fold into per-(query, chunk) key list (s==0 writes, else merges).
      if (lane < 16) {
        unsigned long long* ck = candK + ((size_t)q * NCHNK + g) * KSEL;
        if (s > 0) {
#pragma unroll
          for (int j = 0; j < KSEL; ++j) topk_insert_key(tk, ck[j]);
        }
#pragma unroll
        for (int j = 0; j < KSEL; ++j) ck[j] = tk[j];
      }
    }

    if (w == 0 && s + 1 < NSUBS) __builtin_amdgcn_s_wait_tensorcnt(0);
    __syncthreads();
  }
}

// ---------------- k2: per-query reduce of NCHNK*16 keys -> top-16 indices ----------------
__global__ __launch_bounds__(256) void topk_reduce_kernel(
    const unsigned long long* __restrict__ candK, int* __restrict__ fidx) {
  __shared__ unsigned long long mk[8][32 * KSEL];          // 32KB
  const int w    = threadIdx.x >> 5;
  const int lane = threadIdx.x & 31;
  const int q    = blockIdx.x * 8 + w;                     // one wave per query

  unsigned long long tk[KSEL];
#pragma unroll
  for (int j = 0; j < KSEL; ++j) tk[j] = 0ull;

  const unsigned long long* ck = candK + (size_t)q * (NCHNK * KSEL);
  for (int e = lane; e < NCHNK * KSEL; e += 32) topk_insert_key(tk, ck[e]);

#pragma unroll
  for (int j = 0; j < KSEL; ++j) mk[w][lane * KSEL + j] = tk[j];
  __syncthreads();
  if (lane == 0) {
    unsigned long long fk[KSEL];
#pragma unroll
    for (int j = 0; j < KSEL; ++j) fk[j] = 0ull;
    for (int e = 0; e < 32 * KSEL; ++e) topk_insert_key(fk, mk[w][e]);
#pragma unroll
    for (int j = 0; j < KSEL; ++j)
      fidx[q * KSEL + j] = (int)(unsigned int)(fk[j] & 0xFFFFFFFFull);
  }
}

// ---------------- k3: gather high-res rows + mean pool ----------------
__global__ __launch_bounds__(256) void gather_mean_kernel(
    const float* __restrict__ hires, const int* __restrict__ fidx,
    float* __restrict__ out) {
  const int q  = blockIdx.x;
  const int d4 = threadIdx.x;                              // 256 x float4 = 1024
  float4 acc = make_float4(0.f, 0.f, 0.f, 0.f);
#pragma unroll
  for (int j = 0; j < KSEL; ++j) {
    const int idx = fidx[q * KSEL + j];
    const float4 v = ((const float4*)(hires + (size_t)idx * DHI))[d4];
    acc.x += v.x; acc.y += v.y; acc.z += v.z; acc.w += v.w;
  }
  const float sc = 1.0f / (float)KSEL;
  ((float4*)(out + (size_t)q * DHI))[d4] =
      make_float4(acc.x * sc, acc.y * sc, acc.z * sc, acc.w * sc);
}

// ---------------- host launcher ----------------
extern "C" void kernel_launch(void* const* d_in, const int* in_sizes, int n_in,
                              void* d_out, int out_size, void* d_ws, size_t ws_size,
                              hipStream_t stream) {
  const float* curr  = (const float*)d_in[0];   // [1024, 256]
  const float* dbsat = (const float*)d_in[1];   // [200000, 256]
  const float* dbhi  = (const float*)d_in[2];   // [200000, 1024]
  float* out = (float*)d_out;                   // [1024, 1024]

  char* ws = (char*)d_ws;
  const size_t QBF_BYTES  = (size_t)NB * DSAT * 2;                  // 512 KB
  const size_t DBBF_BYTES = (size_t)NDB * DSAT * 2;                 // 102.4 MB
  const size_t CAND_BYTES = (size_t)NB * NCHNK * KSEL * 8;          // 25.7 MB
  unsigned int*       qbf   = (unsigned int*)ws;
  unsigned int*       dbbf  = (unsigned int*)(ws + QBF_BYTES);
  unsigned long long* candK = (unsigned long long*)(ws + QBF_BYTES + DBBF_BYTES);
  int*                fidx  = (int*)(ws + QBF_BYTES + DBBF_BYTES + CAND_BYTES);

  convert_queries_kernel<<<(NB * DSAT / 2) / 256, 256, 0, stream>>>(curr, qbf);
  convert_db_kernel<<<(NDB * DSAT / 4) / 256, 256, 0, stream>>>(dbsat, dbbf);
  gemm_topk_kernel<<<dim3(2, NCHNK), 256, 0, stream>>>(dbbf, qbf, candK);
  topk_reduce_kernel<<<NB / 8, 256, 0, stream>>>(candK, fidx);
  gather_mean_kernel<<<NB, 256, 0, stream>>>(dbhi, fidx, out);
}
